// InvertedResidualBlock_76794015253060
// MI455X (gfx1250) — matrix operand order repaired and checked by reading, not compiled, not run on hardware
//
#include <hip/hip_runtime.h>

// Fused inverted-residual block for MI455X (gfx1250, wave32, WMMA bf16).
// x:[64,96,56,56] -> pw1(96->576)+BN+ReLU -> dw3x3+BN+ReLU -> pw2(576->96)+BN -> +x

#define EPSV 1e-5f
#define Cin 96
#define HH 56
#define WW 56
#define HIDC 576
#define OUP 96

#define ROWS 2                       // output rows per workgroup
#define HALO_ROWS 4                  // ROWS + 2
#define HALO_COLS 58                 // WW + 2
#define NP_VALID (HALO_ROWS*HALO_COLS)  // 232 halo pixels
#define NP_PAD 240                   // padded to 15 N-tiles of 16
#define XS_STRIDE 104                // 96 ch + pad (avoids LDS bank repeats)
#define H1_STRIDE 240
#define NOUT_PIX (ROWS*WW)           // 112 = 7 N-tiles of 16
#define H2_STRIDE 40                 // 32 ch + pad
#define NTHREADS 192                 // 6 wave32
#define KCHUNKS 18                   // 576 / 32

typedef __attribute__((ext_vector_type(16))) __bf16 v16bf;
typedef __attribute__((ext_vector_type(8)))  __bf16 v8bf;
typedef __attribute__((ext_vector_type(8)))  float  v8f;

// Native conversions: gfx1250 has v_cvt_pk_bf16_f32 / v_cvt_f32_bf16 — let the
// backend emit them instead of a software RNE bit sequence.
__device__ __forceinline__ __bf16 f2bf(float f) { return (__bf16)f; }
__device__ __forceinline__ float  bf2f(__bf16 b) { return (float)b; }

// A-fragment (16x32 bf16, lane m=lane&15, group g=lane>>4 holds
// K = g*16 + [0,16) in order). Scale folds BN gamma/rsqrt(var).
__device__ __forceinline__ v16bf load_a_frag_global(const float* __restrict__ W,
                                                    int ldk, int row, int kb,
                                                    int g, float scale) {
  const float* p = W + row * ldk + kb + g * 16;
  float4 x0 = *(const float4*)(p);
  float4 x1 = *(const float4*)(p + 4);
  float4 x2 = *(const float4*)(p + 8);
  float4 x3 = *(const float4*)(p + 12);
  v16bf a;
  a[0]  = f2bf(x0.x * scale); a[1]  = f2bf(x0.y * scale);
  a[2]  = f2bf(x0.z * scale); a[3]  = f2bf(x0.w * scale);
  a[4]  = f2bf(x1.x * scale); a[5]  = f2bf(x1.y * scale);
  a[6]  = f2bf(x1.z * scale); a[7]  = f2bf(x1.w * scale);
  a[8]  = f2bf(x2.x * scale); a[9]  = f2bf(x2.y * scale);
  a[10] = f2bf(x2.z * scale); a[11] = f2bf(x2.w * scale);
  a[12] = f2bf(x3.x * scale); a[13] = f2bf(x3.y * scale);
  a[14] = f2bf(x3.z * scale); a[15] = f2bf(x3.w * scale);
  return a;
}

// B-fragment (32x16 bf16, contiguous k order, same as A: lane n=lane&15 is the column, group
// g=lane>>4 holds K = g*16 + [0,16) in order). p must point at that K-run.
__device__ __forceinline__ v16bf load_b_frag_lds(const __bf16* p) {
  v8bf lo = *(const v8bf*)(p);
  v8bf hi = *(const v8bf*)(p + 8);
  v16bf b;
#pragma unroll
  for (int j = 0; j < 8; ++j) { b[j] = lo[j]; b[j + 8] = hi[j]; }
  return b;
}

__global__ __launch_bounds__(NTHREADS)
void fused_mbconv_kernel(const float* __restrict__ x,  const float* __restrict__ w1,
                         const float* __restrict__ g1, const float* __restrict__ b1,
                         const float* __restrict__ m1, const float* __restrict__ v1,
                         const float* __restrict__ wd,
                         const float* __restrict__ gd, const float* __restrict__ bd,
                         const float* __restrict__ md, const float* __restrict__ vd,
                         const float* __restrict__ w2,
                         const float* __restrict__ g2, const float* __restrict__ b2,
                         const float* __restrict__ m2, const float* __restrict__ v2,
                         float* __restrict__ out) {
  __shared__ __attribute__((aligned(32))) __bf16 xs[NP_PAD * XS_STRIDE];   // x tile, [pix][ch]
  __shared__ __attribute__((aligned(32))) __bf16 h1s[32 * H1_STRIDE];      // hidden chunk, [ch][pix]
  __shared__ __attribute__((aligned(32))) __bf16 h2s[NOUT_PIX * H2_STRIDE];// dw out, [pix][ch]
  __shared__ float sc1[HIDC], sh1[HIDC], scd[HIDC], shd[HIDC];
  __shared__ float sc2[OUP], sh2[OUP];

  const int tid  = (int)threadIdx.x;
  const int lane = tid & 31;
  const int wid  = tid >> 5;      // 0..5
  const int lm   = lane & 15;
  const int g    = lane >> 4;
  const int bb   = (int)blockIdx.y;            // batch
  const int h0   = (int)blockIdx.x * ROWS;     // first output row

  // ---- fold BatchNorms into per-channel scale/shift (in LDS) ----
  for (int i = tid; i < HIDC; i += NTHREADS) {
    float s1 = g1[i] * rsqrtf(v1[i] + EPSV);
    sc1[i] = s1; sh1[i] = b1[i] - m1[i] * s1;
    float sd = gd[i] * rsqrtf(vd[i] + EPSV);
    scd[i] = sd; shd[i] = bd[i] - md[i] * sd;
  }
  for (int i = tid; i < OUP; i += NTHREADS) {
    float s2 = g2[i] * rsqrtf(v2[i] + EPSV);
    sc2[i] = s2; sh2[i] = b2[i] - m2[i] * s2;
  }
  // zero padded pixel slots 232..239 (read by the last pw1 N-tile)
  for (int i = tid; i < (NP_PAD - NP_VALID) * XS_STRIDE; i += NTHREADS)
    xs[NP_VALID * XS_STRIDE + i] = f2bf(0.f);

  // ---- stage x halo tile (fp32 -> bf16, transposed to [pix][ch]) ----
  for (int idx = tid; idx < NP_VALID * Cin; idx += NTHREADS) {
    int p = idx % NP_VALID;          // pixel fastest -> coalesced global cols
    int c = idx / NP_VALID;
    int r = p / HALO_COLS, cc = p % HALO_COLS;
    int grow = h0 - 1 + r, gcol = cc - 1;
    float fv = 0.f;
    if ((unsigned)grow < (unsigned)HH && (unsigned)gcol < (unsigned)WW)
      fv = x[((bb * Cin + c) * HH + grow) * WW + gcol];
    xs[p * XS_STRIDE + c] = f2bf(fv);
  }
  __syncthreads();

  v8f acc[7] = {};                       // pw2 accumulators: this wave's M-tile x 7 N-tiles
  const int mi = wid & 1;                // pw1 M-tile within chunk
  const int ng = wid >> 1;               // pw1 N-tile group (5 tiles each)
  const int w2row = wid * 16 + lm;       // pw2 output-channel row for A-fragment

  for (int kc = 0; kc < KCHUNKS; ++kc) {
    // ---------- pw1: h1 = ReLU(BN1(W1 @ x)) over halo pixels ----------
    {
      const int arow = kc * 32 + mi * 16 + lm;
      const float s = sc1[arow];
      v16bf a0 = load_a_frag_global(w1, Cin, arow, 0,  g, s);
      v16bf a1 = load_a_frag_global(w1, Cin, arow, 32, g, s);
      v16bf a2 = load_a_frag_global(w1, Cin, arow, 64, g, s);
#pragma unroll
      for (int i = 0; i < 5; ++i) {
        const int nt  = ng * 5 + i;
        const int pix = nt * 16 + lm;
        const __bf16* bp = xs + pix * XS_STRIDE + g * 16;
        v8f d = {};
        d = __builtin_amdgcn_wmma_f32_16x16x32_bf16(false, a0, false, load_b_frag_lds(bp),
                                                    (short)0, d, false, false);
        d = __builtin_amdgcn_wmma_f32_16x16x32_bf16(false, a1, false, load_b_frag_lds(bp + 32),
                                                    (short)0, d, false, false);
        d = __builtin_amdgcn_wmma_f32_16x16x32_bf16(false, a2, false, load_b_frag_lds(bp + 64),
                                                    (short)0, d, false, false);
        // conv pads h (post-BN) with zeros: mask out-of-image halo pixels
        int r = pix / HALO_COLS, cc = pix % HALO_COLS;
        int grow = h0 - 1 + r;
        bool valid = (pix < NP_VALID) && (cc != 0) && (cc != HALO_COLS - 1) &&
                     ((unsigned)grow < (unsigned)HH);
#pragma unroll
        for (int v = 0; v < 8; ++v) {
          int chl = mi * 16 + g * 8 + v;                 // C/D layout: M = v + 8*g
          float val = d[v] + sh1[kc * 32 + chl];
          val = valid ? fmaxf(val, 0.f) : 0.f;
          h1s[chl * H1_STRIDE + pix] = f2bf(val);
        }
      }
    }
    __syncthreads();

    // ---------- depthwise 3x3 + BNd + ReLU (VALU) ----------
    for (int idx = tid; idx < 32 * NOUT_PIX; idx += NTHREADS) {
      int chl = idx / NOUT_PIX;
      int p   = idx - chl * NOUT_PIX;
      int orow = p / WW, ocol = p - orow * WW;
      int kch = kc * 32 + chl;
      const float* wdp = wd + kch * 9;
      const __bf16* hp = h1s + chl * H1_STRIDE + orow * HALO_COLS + ocol;
      float s = 0.f;
#pragma unroll
      for (int dr = 0; dr < 3; ++dr)
#pragma unroll
        for (int dc = 0; dc < 3; ++dc)
          s += wdp[dr * 3 + dc] * bf2f(hp[dr * HALO_COLS + dc]);
      float o = fmaxf(s * scd[kch] + shd[kch], 0.f);
      h2s[p * H2_STRIDE + chl] = f2bf(o);               // [pix][ch] for pw2 B-frags
    }
    __syncthreads();

    // ---------- pw2: accumulate W2 @ h2 into register tiles ----------
    {
      if (kc + 1 < KCHUNKS)
        __builtin_prefetch(wd + (kc + 1) * 32 * 9, 0, 1);   // global_prefetch_b8
      v16bf aw = load_a_frag_global(w2, HIDC, w2row, kc * 32, g, sc2[w2row]);
#pragma unroll
      for (int n = 0; n < 7; ++n) {
        const __bf16* bp = h2s + (n * 16 + lm) * H2_STRIDE + g * 16;
        acc[n] = __builtin_amdgcn_wmma_f32_16x16x32_bf16(false, aw, false, load_b_frag_lds(bp),
                                                         (short)0, acc[n], false, false);
      }
    }
    __syncthreads();
  }

  // ---------- epilogue: BN2 shift + residual + store ----------
#pragma unroll
  for (int n = 0; n < 7; ++n) {
    int p = n * 16 + lm;
    int orow = p / WW, ocol = p - orow * WW;
#pragma unroll
    for (int v = 0; v < 8; ++v) {
      int o  = wid * 16 + g * 8 + v;                    // C/D layout: M = v + 8*g
      int gi = ((bb * OUP + o) * HH + (h0 + orow)) * WW + ocol;
      out[gi] = acc[n][v] + sh2[o] + x[gi];
    }
  }
}

extern "C" void kernel_launch(void* const* d_in, const int* in_sizes, int n_in,
                              void* d_out, int out_size, void* d_ws, size_t ws_size,
                              hipStream_t stream) {
  (void)in_sizes; (void)n_in; (void)d_ws; (void)ws_size; (void)out_size;
  const float* x  = (const float*)d_in[0];
  const float* w1 = (const float*)d_in[1];
  const float* g1 = (const float*)d_in[2];
  const float* b1 = (const float*)d_in[3];
  const float* m1 = (const float*)d_in[4];
  const float* v1 = (const float*)d_in[5];
  const float* wd = (const float*)d_in[6];
  const float* gd = (const float*)d_in[7];
  const float* bd = (const float*)d_in[8];
  const float* md = (const float*)d_in[9];
  const float* vd = (const float*)d_in[10];
  const float* w2 = (const float*)d_in[11];
  const float* g2 = (const float*)d_in[12];
  const float* b2 = (const float*)d_in[13];
  const float* m2 = (const float*)d_in[14];
  const float* v2 = (const float*)d_in[15];
  float* out = (float*)d_out;

  dim3 grid(HH / ROWS, 64);   // 28 row-strips x 64 batch images
  dim3 block(NTHREADS);
  fused_mbconv_kernel<<<grid, block, 0, stream>>>(x, w1, g1, b1, m1, v1,
                                                  wd, gd, bd, md, vd,
                                                  w2, g2, b2, m2, v2, out);
}
